// CombinedLoss_87488483819787
// MI455X (gfx1250) — matrix operand order repaired and checked by reading, not compiled, not run on hardware
//
#include <hip/hip_runtime.h>
#include <math.h>

#define B_ROWS 65536
#define DIM    128
#define M_ANCH 2048
#define HDIM   256
#define INV_TEMP 5.0f
#define TOPK   10

typedef __attribute__((ext_vector_type(16))) __bf16 v16bf;
typedef __attribute__((ext_vector_type(8)))  __bf16 v8bf;
typedef __attribute__((ext_vector_type(8)))  float  v8f;

__device__ __forceinline__ __bf16 f2bf(float f) {
  // round-to-nearest-even f32 -> bf16 (used only in the one-shot prep kernel)
  unsigned u = __builtin_bit_cast(unsigned, f);
  unsigned r = (u + 0x7FFFu + ((u >> 16) & 1u)) >> 16;
  return __builtin_bit_cast(__bf16, (unsigned short)r);
}

__device__ __forceinline__ float wave_reduce_sum(float v) {
#pragma unroll
  for (int o = 16; o > 0; o >>= 1) v += __shfl_xor(v, o, 32);
  return v;
}

// ---------------------------------------------------------------------------
// Kernel 0: one-shot f32 -> bf16 conversion into workspace
// ---------------------------------------------------------------------------
__global__ __launch_bounds__(256) void cvt_bf16_kernel(
    const float* __restrict__ src, __bf16* __restrict__ dst, int n) {
  int i = blockIdx.x * blockDim.x + threadIdx.x;
  int stride = gridDim.x * blockDim.x;
  for (; i < n; i += stride) dst[i] = f2bf(src[i]);
}

// ---------------------------------------------------------------------------
// Kernel 1: fused bf16-WMMA logits + per-row top-10 + logsumexp contrastive
// loss, plus centroid segment-sum scatter and sum(z^2).
// One block = 16 rows; 8 waves each own a 16-column strip per pass.
// ---------------------------------------------------------------------------
__global__ __launch_bounds__(256) void fused_contrastive_centroid_kernel(
    const __bf16* __restrict__ z_bf, const __bf16* __restrict__ anchors_bf,
    const float* __restrict__ z_expr, const int* __restrict__ labels,
    float* __restrict__ acc, float* __restrict__ sums,
    float* __restrict__ counts) {
  __shared__ float s_logits[16 * 128];          // 8 KB staging for one pass
  __shared__ float s_cand[16 * 16 * TOPK];      // 10 KB candidate lists
  __shared__ float s_pos[16];
  __shared__ float s_loss[16];
  __shared__ int   s_label[16];

  const int tid  = threadIdx.x;
  const int wave = tid >> 5;
  const int lane = tid & 31;
  const int row0 = blockIdx.x * 16;

  if (tid < 16) { s_label[tid] = labels[row0 + tid]; s_pos[tid] = 0.0f; }
  __syncthreads();

  // ---- A fragments: 16 rows x 128 K of bf16 z, held in VGPRs whole kernel.
  // ISA 16-bit A layout: lane<16 -> K {0..7,16..23}, lane>=16 -> +8.
  const int ar  = lane & 15;
  const int sel = lane >> 4;
  const __bf16* zrowb = z_bf + (size_t)(row0 + ar) * DIM;
  v16bf afrag[4];
#pragma unroll
  for (int ks = 0; ks < 4; ++ks) {
    v8bf lo = *(const v8bf*)(zrowb + ks * 32 + sel * 8);
    v8bf hi = *(const v8bf*)(zrowb + ks * 32 + sel * 8 + 16);
    afrag[ks] = __builtin_shufflevector(lo, hi, 0, 1, 2, 3, 4, 5, 6, 7,
                                        8, 9, 10, 11, 12, 13, 14, 15);
  }

  // ---- per-thread top-k scan state: 16 threads per row
  const int srow = tid >> 4;  // 0..15 row in strip
  const int scol = tid & 15;  // 0..15 column phase
  const int mylabel = s_label[srow];
  float tk[TOPK];
#pragma unroll
  for (int i = 0; i < TOPK; ++i) tk[i] = -INFINITY;
  float posv = -INFINITY;

  // ---- 16 passes over M=2048 (128 columns per pass)
  for (int p = 0; p < 16; ++p) {
    const int col0 = p * 128 + wave * 16;
    // B layout: column = lane%16, K = ks*32 + sel*8 + {0..7,16..23}  (matches A)
    const __bf16* browp = anchors_bf + (size_t)(col0 + (lane & 15)) * DIM;
    v8f c = {};
#pragma unroll
    for (int ks = 0; ks < 4; ++ks) {
      v8bf blo = *(const v8bf*)(browp + ks * 32 + sel * 8);
      v8bf bhi = *(const v8bf*)(browp + ks * 32 + sel * 8 + 16);
      v16bf bfrag = __builtin_shufflevector(blo, bhi, 0, 1, 2, 3, 4, 5, 6, 7,
                                            8, 9, 10, 11, 12, 13, 14, 15);
      c = __builtin_amdgcn_wmma_f32_16x16x32_bf16(
          false, afrag[ks], false, bfrag, (short)0, c, false, false);
    }
#pragma unroll
    for (int r = 0; r < 8; ++r)
      s_logits[(r + 8 * sel) * 128 + wave * 16 + (lane & 15)] = c[r] * INV_TEMP;
    __syncthreads();

    // scan my 8 columns of this pass
#pragma unroll
    for (int j = 0; j < 8; ++j) {
      int cloc = scol + 16 * j;
      float v  = s_logits[srow * 128 + cloc];
      int gcol = p * 128 + cloc;
      if (gcol == mylabel) {
        posv = v;                         // positive logit, excluded from top-k
      } else if (v > tk[0]) {
        tk[0] = v;                        // bubble-up keeps tk ascending
#pragma unroll
        for (int i = 0; i < TOPK - 1; ++i) {
          float a = tk[i], b = tk[i + 1];
          tk[i]     = fminf(a, b);
          tk[i + 1] = fmaxf(a, b);
        }
      }
    }
    __syncthreads();
  }

  // ---- stage candidates, merge per row, logsumexp
#pragma unroll
  for (int i = 0; i < TOPK; ++i)
    s_cand[(srow * 16 + scol) * TOPK + i] = tk[i];
  if (posv != -INFINITY) s_pos[srow] = posv;
  __syncthreads();

  if (tid < 16) {
    float best[TOPK];
#pragma unroll
    for (int i = 0; i < TOPK; ++i) best[i] = -INFINITY;
    for (int i = 0; i < 16 * TOPK; ++i) {
      float v = s_cand[tid * 16 * TOPK + i];
      if (v > best[0]) {
        best[0] = v;
#pragma unroll
        for (int q = 0; q < TOPK - 1; ++q) {
          float a = best[q], b = best[q + 1];
          best[q]     = fminf(a, b);
          best[q + 1] = fmaxf(a, b);
        }
      }
    }
    float pv = s_pos[tid];
    float mx = pv;
#pragma unroll
    for (int i = 0; i < TOPK; ++i) mx = fmaxf(mx, best[i]);
    float se = __expf(pv - mx);
#pragma unroll
    for (int i = 0; i < TOPK; ++i) se += __expf(best[i] - mx);
    s_loss[tid] = mx + __logf(se) - pv;
  }
  __syncthreads();
  if (tid == 0) {
    float t = 0.0f;
#pragma unroll
    for (int i = 0; i < 16; ++i) t += s_loss[i];
    atomicAdd(&acc[0], t);
  }

  // ---- centroid segment sums + sum(z^2)  (fp32 source for accuracy)
  const int lab = s_label[srow];
  const float* zr2 = z_expr + (size_t)(row0 + srow) * DIM;
  float sq = 0.0f;
#pragma unroll
  for (int j = 0; j < 8; ++j) {
    int d   = scol + 16 * j;
    float z = zr2[d];
    sq += z * z;
    atomicAdd(&sums[(size_t)lab * DIM + d], z);
  }
  sq = wave_reduce_sum(sq);
  if (lane == 0) atomicAdd(&acc[1], sq);
  if (tid < 16) atomicAdd(&counts[s_label[tid]], 1.0f);
}

// ---------------------------------------------------------------------------
// Kernel 2: h-space alignment sum of squared diffs (memory-bound, float4)
// ---------------------------------------------------------------------------
__global__ __launch_bounds__(256) void halign_kernel(
    const float* __restrict__ he, const float* __restrict__ hc,
    float* __restrict__ acc2) {
  const size_t n4 = (size_t)B_ROWS * HDIM / 4;
  const float4* a = (const float4*)he;
  const float4* b = (const float4*)hc;
  size_t i = (size_t)blockIdx.x * blockDim.x + threadIdx.x;
  const size_t stride = (size_t)gridDim.x * blockDim.x;
  float s = 0.0f;
  for (; i < n4; i += stride) {
    float4 x = a[i], y = b[i];
    float d0 = x.x - y.x, d1 = x.y - y.y, d2 = x.z - y.z, d3 = x.w - y.w;
    s += d0 * d0 + d1 * d1 + d2 * d2 + d3 * d3;
  }
  s = wave_reduce_sum(s);
  if ((threadIdx.x & 31) == 0) atomicAdd(acc2, s);
}

// ---------------------------------------------------------------------------
// Kernel 3: sum_m ||sums_m||^2 / max(count_m, 1)
// ---------------------------------------------------------------------------
__global__ __launch_bounds__(256) void centroid_term_kernel(
    const float* __restrict__ sums, const float* __restrict__ counts,
    float* __restrict__ acc3) {
  int m = blockIdx.x * blockDim.x + threadIdx.x;
  float s = 0.0f;
  if (m < M_ANCH) {
    const float* p = sums + (size_t)m * DIM;
#pragma unroll 4
    for (int d = 0; d < DIM; ++d) s += p[d] * p[d];
    s /= fmaxf(counts[m], 1.0f);
  }
  s = wave_reduce_sum(s);
  if ((threadIdx.x & 31) == 0) atomicAdd(acc3, s);
}

// ---------------------------------------------------------------------------
// Kernel 4: combine partial accumulators into the scalar loss
// ---------------------------------------------------------------------------
__global__ void finalize_kernel(const float* __restrict__ acc,
                                float* __restrict__ out) {
  if (threadIdx.x == 0 && blockIdx.x == 0) {
    float contrastive = acc[0] / (float)B_ROWS;
    float centroid    = (acc[1] - acc[3]) / ((float)B_ROWS * (float)DIM);
    float halign      = acc[2] / ((float)B_ROWS * (float)HDIM);
    out[0] = contrastive + 0.05f * centroid + 0.1f * halign;
  }
}

extern "C" void kernel_launch(void* const* d_in, const int* in_sizes, int n_in,
                              void* d_out, int out_size, void* d_ws, size_t ws_size,
                              hipStream_t stream) {
  const float* z_expr  = (const float*)d_in[0];
  const float* h_expr  = (const float*)d_in[1];
  const float* h_cnv   = (const float*)d_in[2];
  const float* anchors = (const float*)d_in[3];
  const int*   labels  = (const int*)d_in[4];

  // workspace layout (all offsets 64B-aligned)
  char* ws_base = (char*)d_ws;
  float*  acc        = (float*)ws_base;                               // 16 floats
  float*  sums       = (float*)(ws_base + 64);                        // M*D floats
  float*  counts     = (float*)(ws_base + 64 + (size_t)M_ANCH * DIM * 4);   // M floats
  __bf16* z_bf       = (__bf16*)(ws_base + 64 + (size_t)M_ANCH * DIM * 4 +
                                 (size_t)M_ANCH * 4);                 // B*D bf16
  __bf16* anchors_bf = (__bf16*)((char*)z_bf + (size_t)B_ROWS * DIM * 2); // M*D bf16

  const size_t zero_bytes = 64 + (size_t)M_ANCH * DIM * 4 + (size_t)M_ANCH * 4;
  hipMemsetAsync(d_ws, 0, zero_bytes, stream);

  cvt_bf16_kernel<<<2048, 256, 0, stream>>>(z_expr, z_bf, B_ROWS * DIM);
  cvt_bf16_kernel<<<256, 256, 0, stream>>>(anchors, anchors_bf, M_ANCH * DIM);

  fused_contrastive_centroid_kernel<<<B_ROWS / 16, 256, 0, stream>>>(
      z_bf, anchors_bf, z_expr, labels, acc, sums, counts);
  halign_kernel<<<1024, 256, 0, stream>>>(h_expr, h_cnv, &acc[2]);
  centroid_term_kernel<<<M_ANCH / 256, 256, 0, stream>>>(sums, counts, &acc[3]);
  finalize_kernel<<<1, 32, 0, stream>>>(acc, (float*)d_out);
}
